// UnifiedLoss_90890097918700
// MI455X (gfx1250) — compile-verified, matrix operand order checked by reading, not run on hardware
//
#include <hip/hip_runtime.h>
#include <hip/hip_bf16.h>

typedef _Float16 half_t;
typedef __attribute__((ext_vector_type(16))) _Float16 v16h;
typedef __attribute__((ext_vector_type(8)))  _Float16 v8h;
typedef __attribute__((ext_vector_type(8)))  float    v8f;

#define D_DIM 512
#define C_CLS 9
#define B_BATCH 32
#define BN 2048
#define NTOT (B_BATCH * BN)   /* 65536 flattened rows */

// ---------------------------------------------------------------------------
// WMMA fragment loader (CDNA5 16-bit A 16x32 layout; B = columns of W, same
// per-lane pattern since W is stored (d_out, k) row-major and B col N = W row N)
// lane L in [0,16): row/col = base+L, chunks K{8h..8h+7} and K{16+8h..16+8h+7}
// ---------------------------------------------------------------------------
__device__ __forceinline__ v16h frag_load_f16(const half_t* __restrict__ base,
                                              int row_base, int k_base) {
  const int lane = threadIdx.x & 31;
  const int L = lane & 15;
  const int h = lane >> 4;
  const half_t* p = base + (size_t)(row_base + L) * D_DIM + k_base;
  v8h a = *reinterpret_cast<const v8h*>(p + 8 * h);
  v8h b = *reinterpret_cast<const v8h*>(p + 16 + 8 * h);
  v16h f;
#pragma unroll
  for (int j = 0; j < 8; ++j) { f[j] = a[j]; f[8 + j] = b[j]; }
  return f;
}

// ---------------------------------------------------------------------------
// Complex GEMM: Y = X @ W^T + bias (complex), f16 WMMA, f32 accumulate.
//   Yr = Xr*Wr + Xi*(-Wi) ; Yi = Xr*Wi + Xi*Wr   (Wineg pre-negated: f16 WMMA
//   only allows C-negation per ISA NEG table, so A/B negation is baked in)
// MODE 0: store f16 (bias added).  MODE 1: residual blend with sigmoid(alpha).
// Block = 256 threads = 8 waves (4 row-waves x 2 col-waves).
// Wave tile = 32x32 complex (2x2 WMMA sub-tiles): 10 fragment loads per
// 16 WMMAs per k-step -> ~26 FLOP/byte from memory.
// Block tile = 128 rows x 64 cols.
// ---------------------------------------------------------------------------
template <int MODE>
__global__ __launch_bounds__(256)
void cgemm_wmma(const half_t* __restrict__ Ar, const half_t* __restrict__ Ai,
                const half_t* __restrict__ Wr, const half_t* __restrict__ Wi,
                const half_t* __restrict__ Wineg,
                const float* __restrict__ br, const float* __restrict__ bi,
                half_t* __restrict__ Cr, half_t* __restrict__ Ci,
                const float* __restrict__ blendR, const float* __restrict__ blendI,
                const float* __restrict__ alphaPtr) {
  const int wave = threadIdx.x >> 5;
  const int row0 = blockIdx.x * 128 + (wave & 3) * 32;
  const int col0 = blockIdx.y * 64 + (wave >> 2) * 32;

  v8f accR[2][2] = {};
  v8f accI[2][2] = {};

  // lane l prefetches row row0+l -> 32 lanes cover the whole 32-row wave tile
  const half_t* prefA = Ar + (size_t)(row0 + (threadIdx.x & 31)) * D_DIM;

#pragma unroll 1
  for (int k = 0; k < D_DIM; k += 32) {
    if (k + 32 < D_DIM) __builtin_prefetch(prefA + k + 32, 0, 1);  // global_prefetch_b8
    v16h xr[2], xi[2], wr[2], wi[2], wn[2];
#pragma unroll
    for (int rt = 0; rt < 2; ++rt) {
      xr[rt] = frag_load_f16(Ar, row0 + rt * 16, k);
      xi[rt] = frag_load_f16(Ai, row0 + rt * 16, k);
    }
#pragma unroll
    for (int ct = 0; ct < 2; ++ct) {
      wr[ct] = frag_load_f16(Wr,    col0 + ct * 16, k);
      wi[ct] = frag_load_f16(Wi,    col0 + ct * 16, k);
      wn[ct] = frag_load_f16(Wineg, col0 + ct * 16, k);
    }
#pragma unroll
    for (int rt = 0; rt < 2; ++rt)
#pragma unroll
      for (int ct = 0; ct < 2; ++ct) {
        accR[rt][ct] = __builtin_amdgcn_wmma_f32_16x16x32_f16(
            false, xr[rt], false, wr[ct], (short)0, accR[rt][ct], false, false);
        accR[rt][ct] = __builtin_amdgcn_wmma_f32_16x16x32_f16(
            false, xi[rt], false, wn[ct], (short)0, accR[rt][ct], false, false);
        accI[rt][ct] = __builtin_amdgcn_wmma_f32_16x16x32_f16(
            false, xr[rt], false, wi[ct], (short)0, accI[rt][ct], false, false);
        accI[rt][ct] = __builtin_amdgcn_wmma_f32_16x16x32_f16(
            false, xi[rt], false, wr[ct], (short)0, accI[rt][ct], false, false);
      }
  }

  // C/D layout: lane = L + 16h -> col = base + L; element v -> row base + v + 8h
  const int lane = threadIdx.x & 31;
  const int L = lane & 15;
  const int h = lane >> 4;
  float alpha = 0.f, oneMinus = 0.f;
  if (MODE == 1) {
    const float pa = alphaPtr[0];
    alpha = 1.f / (1.f + expf(-pa));
    oneMinus = 1.f - alpha;
  }
#pragma unroll
  for (int ct = 0; ct < 2; ++ct) {
    const int col = col0 + ct * 16 + L;
    const float bR = br[col], bI = bi[col];
#pragma unroll
    for (int rt = 0; rt < 2; ++rt) {
#pragma unroll
      for (int v = 0; v < 8; ++v) {
        const int row = row0 + rt * 16 + v + 8 * h;
        const size_t o = (size_t)row * D_DIM + col;
        float yr = accR[rt][ct][v] + bR;
        float yi = accI[rt][ct][v] + bI;
        if (MODE == 1) {
          yr = alpha * yr + oneMinus * blendR[o];
          yi = alpha * yi + oneMinus * blendI[o];
        }
        Cr[o] = (half_t)yr;
        Ci[o] = (half_t)yi;
      }
    }
  }
}

// ---------------------------------------------------------------------------
// Block reductions (256 threads)
// ---------------------------------------------------------------------------
__device__ __forceinline__ float block_sum_256(float v, float* sbuf) {
  const int tid = threadIdx.x;
  sbuf[tid] = v;
  __syncthreads();
#pragma unroll
  for (int s = 128; s > 0; s >>= 1) {
    if (tid < s) sbuf[tid] += sbuf[tid + s];
    __syncthreads();
  }
  const float r = sbuf[0];
  __syncthreads();
  return r;
}

__device__ __forceinline__ float block_max_256(float v, float* sbuf) {
  const int tid = threadIdx.x;
  sbuf[tid] = v;
  __syncthreads();
#pragma unroll
  for (int s = 128; s > 0; s >>= 1) {
    if (tid < s) sbuf[tid] = fmaxf(sbuf[tid], sbuf[tid + s]);
    __syncthreads();
  }
  const float r = sbuf[0];
  __syncthreads();
  return r;
}

// ---------------------------------------------------------------------------
// Complex LayerNorm (2x2 covariance whitening) + CPReLU, one block per row
// ---------------------------------------------------------------------------
__global__ __launch_bounds__(256)
void ln_prelu(const half_t* __restrict__ Yr, const half_t* __restrict__ Yi,
              half_t* __restrict__ Zr, half_t* __restrict__ Zi,
              const float* __restrict__ grr, const float* __restrict__ gii,
              const float* __restrict__ gri, const float* __restrict__ betar,
              const float* __restrict__ betai,
              const float* __restrict__ arPtr, const float* __restrict__ aiPtr) {
  __shared__ float sbuf[256];
  const int row = blockIdx.x;
  const int tid = threadIdx.x;
  const size_t base = (size_t)row * D_DIM;
  const int c[2] = { tid * 2, tid * 2 + 1 };
  float r[2], im[2];
#pragma unroll
  for (int e = 0; e < 2; ++e) { r[e] = (float)Yr[base + c[e]]; im[e] = (float)Yi[base + c[e]]; }

  const float mr = block_sum_256(r[0] + r[1], sbuf) * (1.f / 512.f);
  const float mi = block_sum_256(im[0] + im[1], sbuf) * (1.f / 512.f);
#pragma unroll
  for (int e = 0; e < 2; ++e) { r[e] -= mr; im[e] -= mi; }

  const float vrr = block_sum_256(r[0] * r[0] + r[1] * r[1], sbuf) * (1.f / 512.f) + 1e-5f;
  const float vii = block_sum_256(im[0] * im[0] + im[1] * im[1], sbuf) * (1.f / 512.f) + 1e-5f;
  const float vri = block_sum_256(r[0] * im[0] + r[1] * im[1], sbuf) * (1.f / 512.f);

  const float s = sqrtf(vrr * vii - vri * vri);
  const float t = sqrtf(vrr + vii + 2.f * s);
  const float inv = 1.f / (s * t);
  const float wrr = (vii + s) * inv;
  const float wii = (vrr + s) * inv;
  const float wri = -vri * inv;
  const float par = arPtr[0], pai = aiPtr[0];

#pragma unroll
  for (int e = 0; e < 2; ++e) {
    const int cc = c[e];
    const float yr = wrr * r[e] + wri * im[e];
    const float yi = wri * r[e] + wii * im[e];
    float orr = grr[cc] * yr + gri[cc] * yi + betar[cc];
    float oii = gri[cc] * yr + gii[cc] * yi + betai[cc];
    orr = orr >= 0.f ? orr : par * orr;
    oii = oii >= 0.f ? oii : pai * oii;
    Zr[base + cc] = (half_t)orr;
    Zi[base + cc] = (half_t)oii;
  }
}

// ---------------------------------------------------------------------------
// q = cv_linear(prototypes) -> qws = [qr(9*512), qi(9*512)] fp32
// ---------------------------------------------------------------------------
__global__ __launch_bounds__(512)
void q_linear(const float* __restrict__ protos, const float* __restrict__ Wr,
              const float* __restrict__ Wi, const float* __restrict__ br,
              const float* __restrict__ bi, float* __restrict__ qws) {
  __shared__ float spr[D_DIM], spi[D_DIM];
  const int c = blockIdx.x;
  const int dout = threadIdx.x;
  spr[dout] = protos[c * D_DIM + dout];
  spi[dout] = protos[C_CLS * D_DIM + c * D_DIM + dout];
  __syncthreads();
  float yr = br[dout], yi = bi[dout];
  const float* wr = Wr + (size_t)dout * D_DIM;
  const float* wi = Wi + (size_t)dout * D_DIM;
  for (int k = 0; k < D_DIM; ++k) {
    const float pr = spr[k], pi = spi[k], wrv = wr[k], wiv = wi[k];
    yr += pr * wrv - pi * wiv;
    yi += pr * wiv + pi * wrv;
  }
  qws[c * D_DIM + dout] = yr;
  qws[C_CLS * D_DIM + c * D_DIM + dout] = yi;
}

// ---------------------------------------------------------------------------
// scores[b,c,n] = (qr[c].Kr[b,n] + qi[c].Ki[b,n]) * head_dim^-0.5 (=0.125)
// one thread per (b,n); q staged in LDS (36 KB)
// ---------------------------------------------------------------------------
__global__ __launch_bounds__(256)
void scores_kernel(const half_t* __restrict__ Kr, const half_t* __restrict__ Ki,
                   const float* __restrict__ qws, float* __restrict__ scores) {
  __shared__ float sq[2 * C_CLS * D_DIM];
  const int tid = threadIdx.x;
  for (int i = tid; i < 2 * C_CLS * D_DIM; i += 256) sq[i] = qws[i];
  __syncthreads();
  const int idx = blockIdx.x * 256 + tid;
  const int b = idx >> 11;
  const int n = idx & (BN - 1);
  float acc[C_CLS] = {};
  const v8h* kr8 = reinterpret_cast<const v8h*>(Kr + (size_t)idx * D_DIM);
  const v8h* ki8 = reinterpret_cast<const v8h*>(Ki + (size_t)idx * D_DIM);
  for (int d8 = 0; d8 < D_DIM / 8; ++d8) {
    const v8h kr = kr8[d8];
    const v8h ki = ki8[d8];
#pragma unroll
    for (int j = 0; j < 8; ++j) {
      const int d = d8 * 8 + j;
      const float krf = (float)kr[j];
      const float kif = (float)ki[j];
#pragma unroll
      for (int c = 0; c < C_CLS; ++c)
        acc[c] += sq[c * D_DIM + d] * krf + sq[C_CLS * D_DIM + c * D_DIM + d] * kif;
    }
  }
#pragma unroll
  for (int c = 0; c < C_CLS; ++c)
    scores[((size_t)(b * C_CLS + c)) * BN + n] = acc[c] * 0.125f;
}

// ---------------------------------------------------------------------------
// Per-(b,c) row: standardize (ddof=1) * softplus(gain), softmax over n
// ---------------------------------------------------------------------------
__global__ __launch_bounds__(256)
void softmax_norm(const float* __restrict__ scores, float* __restrict__ attn,
                  const float* __restrict__ gainPtr) {
  __shared__ float sbuf[256];
  const int row = blockIdx.x;  // b*C + c
  const int tid = threadIdx.x;
  const float* src = scores + (size_t)row * BN;
  float* dst = attn + (size_t)row * BN;
  float x[8];
  float ssum = 0.f, ssq = 0.f;
#pragma unroll
  for (int j = 0; j < 8; ++j) {
    x[j] = src[tid + 256 * j];
    ssum += x[j];
    ssq += x[j] * x[j];
  }
  const float tot  = block_sum_256(ssum, sbuf);
  const float tot2 = block_sum_256(ssq, sbuf);
  const float mean = tot * (1.f / 2048.f);
  float var = (tot2 - 2048.f * mean * mean) * (1.f / 2047.f);
  var = var > 0.f ? var : 0.f;
  const float stdv = sqrtf(var) + 1e-6f;
  const float g = gainPtr[0];
  const float sp = g > 20.f ? g : log1pf(expf(g));
  const float kf = sp / stdv;
#pragma unroll
  for (int j = 0; j < 8; ++j) x[j] = (x[j] - mean) * kf;
  float mx = x[0];
#pragma unroll
  for (int j = 1; j < 8; ++j) mx = fmaxf(mx, x[j]);
  const float bmx = block_max_256(mx, sbuf);
  float es = 0.f;
#pragma unroll
  for (int j = 0; j < 8; ++j) { x[j] = expf(x[j] - bmx); es += x[j]; }
  const float tes = block_sum_256(es, sbuf);
  const float invs = 1.f / tes;
#pragma unroll
  for (int j = 0; j < 8; ++j) dst[tid + 256 * j] = x[j] * invs;
}

// ---------------------------------------------------------------------------
// h[b,c,d] = sum_n attn[b,c,n] * V[b,n,d]   (one block per b, thread = d)
// ---------------------------------------------------------------------------
__global__ __launch_bounds__(512)
void attn_v(const float* __restrict__ attn, const half_t* __restrict__ Vr,
            const half_t* __restrict__ Vi, float* __restrict__ hR,
            float* __restrict__ hI) {
  __shared__ float sat[C_CLS * 128];
  const int b = blockIdx.x;
  const int d = threadIdx.x;
  float aR[C_CLS] = {};
  float aI[C_CLS] = {};
  for (int n0 = 0; n0 < BN; n0 += 128) {
    __syncthreads();
    for (int i = d; i < C_CLS * 128; i += 512) {
      const int c = i >> 7, nn = i & 127;
      sat[i] = attn[((size_t)(b * C_CLS + c)) * BN + n0 + nn];
    }
    __syncthreads();
    for (int nn = 0; nn < 128; ++nn) {
      const size_t o = ((size_t)(b * BN + n0 + nn)) * D_DIM + d;
      const float vr = (float)Vr[o];
      const float vi = (float)Vi[o];
#pragma unroll
      for (int c = 0; c < C_CLS; ++c) {
        const float a = sat[(c << 7) + nn];
        aR[c] += a * vr;
        aI[c] += a * vi;
      }
    }
  }
#pragma unroll
  for (int c = 0; c < C_CLS; ++c) {
    hR[((size_t)(b * C_CLS + c)) * D_DIM + d] = aR[c];
    hI[((size_t)(b * C_CLS + c)) * D_DIM + d] = aI[c];
  }
}

// ---------------------------------------------------------------------------
// out linear (complex, fp32 — tiny): one block per (b,c)
// ---------------------------------------------------------------------------
__global__ __launch_bounds__(256)
void out_linear(const float* __restrict__ hR, const float* __restrict__ hI,
                const float* __restrict__ Wr, const float* __restrict__ Wi,
                const float* __restrict__ br, const float* __restrict__ bi,
                float* __restrict__ oR, float* __restrict__ oI) {
  __shared__ float shr[D_DIM], shi[D_DIM];
  const int bc = blockIdx.x;
  const int tid = threadIdx.x;
  for (int i = tid; i < D_DIM; i += 256) {
    shr[i] = hR[(size_t)bc * D_DIM + i];
    shi[i] = hI[(size_t)bc * D_DIM + i];
  }
  __syncthreads();
  for (int du = 0; du < 2; ++du) {
    const int dout = tid + du * 256;
    float yr = br[dout], yi = bi[dout];
    const float* wr = Wr + (size_t)dout * D_DIM;
    const float* wi = Wi + (size_t)dout * D_DIM;
    for (int k = 0; k < D_DIM; ++k) {
      const float hr = shr[k], hh = shi[k];
      const float wrv = wr[k], wiv = wi[k];
      yr += hr * wrv - hh * wiv;
      yi += hr * wiv + hh * wrv;
    }
    oR[(size_t)bc * D_DIM + dout] = yr;
    oI[(size_t)bc * D_DIM + dout] = yi;
  }
}

// ---------------------------------------------------------------------------
// Prototype distance -> logits -> log_softmax, per-b loss + argmax
// ---------------------------------------------------------------------------
__global__ __launch_bounds__(256)
void proto_loss(const float* __restrict__ oR, const float* __restrict__ oI,
                const float* __restrict__ protos, const int* __restrict__ labels,
                float* __restrict__ lossv, float* __restrict__ outbuf) {
  __shared__ float sbuf[256];
  __shared__ float d2s[C_CLS];
  const int b = blockIdx.x;
  const int tid = threadIdx.x;
  for (int c = 0; c < C_CLS; ++c) {
    float p = 0.f;
    for (int d = tid; d < D_DIM; d += 256) {
      const float dr = oR[((size_t)(b * C_CLS + c)) * D_DIM + d] - protos[c * D_DIM + d];
      const float di = oI[((size_t)(b * C_CLS + c)) * D_DIM + d] - protos[C_CLS * D_DIM + c * D_DIM + d];
      p += dr * dr + di * di;
    }
    const float tot = block_sum_256(p, sbuf);
    if (tid == 0) d2s[c] = tot;
    __syncthreads();
  }
  if (tid == 0) {
    float logits[C_CLS];
    float mx = -3.4e38f;
    for (int c = 0; c < C_CLS; ++c) {
      logits[c] = -d2s[c];
      mx = fmaxf(mx, logits[c]);
    }
    float se = 0.f;
    for (int c = 0; c < C_CLS; ++c) se += expf(logits[c] - mx);
    const float lse = mx + logf(se);
    const int lab = labels[b];
    lossv[b] = -(logits[lab] - lse);
    int am = 0;
    float best = logits[0];
    for (int c = 1; c < C_CLS; ++c)
      if (logits[c] > best) { best = logits[c]; am = c; }
    outbuf[1 + b] = (float)am;
  }
}

__global__ void loss_reduce(const float* __restrict__ lossv, float* __restrict__ outbuf) {
  if (threadIdx.x == 0 && blockIdx.x == 0) {
    float s = 0.f;
    for (int b = 0; b < B_BATCH; ++b) s += lossv[b];
    outbuf[0] = s * (1.f / (float)B_BATCH);
  }
}

// Weight convert: fp32 -> f16, plus pre-negated Wi for the Yr accumulation
__global__ __launch_bounds__(256)
void convert_w(const float* __restrict__ wr, const float* __restrict__ wi,
               half_t* __restrict__ hr, half_t* __restrict__ hi,
               half_t* __restrict__ hineg) {
  const int i = blockIdx.x * 256 + threadIdx.x;
  hr[i] = (half_t)wr[i];
  const float w = wi[i];
  hi[i] = (half_t)w;
  hineg[i] = (half_t)(-w);
}

// Activation convert: fp32 plane -> f16 plane, 8 elements per thread (b128 in,
// b128 out) so the GEMM inner loops stay pure load+wmma.
__global__ __launch_bounds__(256)
void convert_act(const float* __restrict__ src, half_t* __restrict__ dst) {
  const size_t i = ((size_t)blockIdx.x * 256 + threadIdx.x) * 8;
  v8f a = *reinterpret_cast<const v8f*>(src + i);
  v8h o;
#pragma unroll
  for (int j = 0; j < 8; ++j) o[j] = (_Float16)a[j];
  *reinterpret_cast<v8h*>(dst + i) = o;
}

// ---------------------------------------------------------------------------
// Host launcher. Input order = recursive insertion-order flatten of
// setup_inputs(): band_real, band_imag, labels, prototypes, proj_alpha,
// attn_gain, lin1{wr,wi,br,bi}, ln{grr,gii,gri,betar,betai}, prelu{ar,ai},
// lin2{wr,wi,br,bi}, q{..}, k{..}, v{..}, out{..}
// ---------------------------------------------------------------------------
extern "C" void kernel_launch(void* const* d_in, const int* in_sizes, int n_in,
                              void* d_out, int out_size, void* d_ws, size_t ws_size,
                              hipStream_t stream) {
  (void)in_sizes; (void)n_in; (void)out_size; (void)ws_size;

  const float* bandR  = (const float*)d_in[0];
  const float* bandI  = (const float*)d_in[1];
  const int*   labels = (const int*)d_in[2];
  const float* protos = (const float*)d_in[3];
  const float* alphaP = (const float*)d_in[4];
  const float* gainP  = (const float*)d_in[5];
  const float* l1wr = (const float*)d_in[6],  *l1wi = (const float*)d_in[7];
  const float* l1br = (const float*)d_in[8],  *l1bi = (const float*)d_in[9];
  const float* grr  = (const float*)d_in[10], *gii  = (const float*)d_in[11];
  const float* gri  = (const float*)d_in[12];
  const float* betar = (const float*)d_in[13], *betai = (const float*)d_in[14];
  const float* pAr = (const float*)d_in[15], *pAi = (const float*)d_in[16];
  const float* l2wr = (const float*)d_in[17], *l2wi = (const float*)d_in[18];
  const float* l2br = (const float*)d_in[19], *l2bi = (const float*)d_in[20];
  const float* qwr = (const float*)d_in[21], *qwi = (const float*)d_in[22];
  const float* qbr = (const float*)d_in[23], *qbi = (const float*)d_in[24];
  const float* kwr = (const float*)d_in[25], *kwi = (const float*)d_in[26];
  const float* kbr = (const float*)d_in[27], *kbi = (const float*)d_in[28];
  const float* vwr = (const float*)d_in[29], *vwi = (const float*)d_in[30];
  const float* vbr = (const float*)d_in[31], *vbi = (const float*)d_in[32];
  const float* owr = (const float*)d_in[33], *owi = (const float*)d_in[34];
  const float* obr = (const float*)d_in[35], *obi = (const float*)d_in[36];

  // ---- workspace carving ----
  char* ws = (char*)d_ws;
  size_t off = 0;
  auto carve = [&](size_t bytes) -> void* {
    void* p = ws + off;
    off += (bytes + 255) & ~(size_t)255;
    return p;
  };
  const size_t actE = (size_t)NTOT * D_DIM;          // activation plane elems
  half_t* Bhr = (half_t*)carve(actE * 2);            // band f16 (later reused for V)
  half_t* Bhi = (half_t*)carve(actE * 2);
  half_t* A0r = (half_t*)carve(actE * 2);
  half_t* A0i = (half_t*)carve(actE * 2);
  half_t* A1r = (half_t*)carve(actE * 2);
  half_t* A1i = (half_t*)carve(actE * 2);
  const size_t wE = (size_t)D_DIM * D_DIM;
  struct WH { half_t *r, *i, *in; };
  WH w1 = { (half_t*)carve(wE * 2), (half_t*)carve(wE * 2), (half_t*)carve(wE * 2) };
  WH w2 = { (half_t*)carve(wE * 2), (half_t*)carve(wE * 2), (half_t*)carve(wE * 2) };
  WH wk = { (half_t*)carve(wE * 2), (half_t*)carve(wE * 2), (half_t*)carve(wE * 2) };
  WH wv = { (half_t*)carve(wE * 2), (half_t*)carve(wE * 2), (half_t*)carve(wE * 2) };
  float* qws    = (float*)carve((size_t)2 * C_CLS * D_DIM * 4);
  float* scores = (float*)carve((size_t)B_BATCH * C_CLS * BN * 4);
  float* attn   = (float*)carve((size_t)B_BATCH * C_CLS * BN * 4);
  float* hR = (float*)carve((size_t)B_BATCH * C_CLS * D_DIM * 4);
  float* hI = (float*)carve((size_t)B_BATCH * C_CLS * D_DIM * 4);
  float* oR = (float*)carve((size_t)B_BATCH * C_CLS * D_DIM * 4);
  float* oI = (float*)carve((size_t)B_BATCH * C_CLS * D_DIM * 4);
  float* lossv = (float*)carve(B_BATCH * 4);
  float* outF = (float*)d_out;

  // 1) weights -> f16 (+negated Wi); band -> f16 planes
  const dim3 gW(wE / 256);
  convert_w<<<gW, 256, 0, stream>>>(l1wr, l1wi, w1.r, w1.i, w1.in);
  convert_w<<<gW, 256, 0, stream>>>(l2wr, l2wi, w2.r, w2.i, w2.in);
  convert_w<<<gW, 256, 0, stream>>>(kwr, kwi, wk.r, wk.i, wk.in);
  convert_w<<<gW, 256, 0, stream>>>(vwr, vwi, wv.r, wv.i, wv.in);
  const dim3 gA(actE / (256 * 8));
  convert_act<<<gA, 256, 0, stream>>>(bandR, Bhr);
  convert_act<<<gA, 256, 0, stream>>>(bandI, Bhi);

  const dim3 gG(NTOT / 128, D_DIM / 64);
  // 2) lin1: Bh -> A0
  cgemm_wmma<0><<<gG, 256, 0, stream>>>(Bhr, Bhi, w1.r, w1.i, w1.in,
                                        l1br, l1bi, A0r, A0i,
                                        nullptr, nullptr, nullptr);
  // 3) complex LN + PReLU: A0 -> A1
  ln_prelu<<<NTOT, 256, 0, stream>>>(A0r, A0i, A1r, A1i, grr, gii, gri, betar,
                                     betai, pAr, pAi);
  // 4) lin2 + sigmoid(alpha) residual blend with original fp32 band: A1 -> A0
  cgemm_wmma<1><<<gG, 256, 0, stream>>>(A1r, A1i, w2.r, w2.i, w2.in,
                                        l2br, l2bi, A0r, A0i,
                                        bandR, bandI, alphaP);
  // 5) K = k(blended): A0 -> A1
  cgemm_wmma<0><<<gG, 256, 0, stream>>>(A0r, A0i, wk.r, wk.i, wk.in,
                                        kbr, kbi, A1r, A1i,
                                        nullptr, nullptr, nullptr);
  // 6) q on prototypes (tiny, fp32)
  q_linear<<<C_CLS, 512, 0, stream>>>(protos, qwr, qwi, qbr, qbi, qws);
  // 7) scores (consumes K)
  scores_kernel<<<NTOT / 256, 256, 0, stream>>>(A1r, A1i, qws, scores);
  // 8) standardize + softplus gain + softmax
  softmax_norm<<<B_BATCH * C_CLS, 256, 0, stream>>>(scores, attn, gainP);
  // 9) V = v(blended): A0 -> Bh (band f16 planes no longer needed)
  cgemm_wmma<0><<<gG, 256, 0, stream>>>(A0r, A0i, wv.r, wv.i, wv.in,
                                        vbr, vbi, Bhr, Bhi,
                                        nullptr, nullptr, nullptr);
  // 10) h = attn @ V
  attn_v<<<B_BATCH, 512, 0, stream>>>(attn, Bhr, Bhi, hR, hI);
  // 11) out linear
  out_linear<<<B_BATCH * C_CLS, 256, 0, stream>>>(hR, hI, owr, owi, obr, obi, oR, oI);
  // 12) prototype distance loss + preds
  proto_loss<<<B_BATCH, 256, 0, stream>>>(oR, oI, protos, labels, lossv, outF);
  // 13) mean loss
  loss_reduce<<<1, 32, 0, stream>>>(lossv, outF);
}
